// CausalMultiheadSelfAttention_22282290331680
// MI455X (gfx1250) — compile-verified, matrix operand order checked by reading, not run on hardware
//
#include <hip/hip_runtime.h>

// CDNA5 / gfx1250: wave32, WMMA 16x16x32 bf16 with fp32 accumulate.
typedef __attribute__((ext_vector_type(16))) __bf16 v16bf;
typedef __attribute__((ext_vector_type(8)))  __bf16 v8bf;
typedef __attribute__((ext_vector_type(8)))  float  v8f;

#define B_  2
#define S_  2048
#define D_  1024
#define H_  16
#define DK_ 64
#define N_  (B_ * S_)   // 4096 token rows

__device__ __forceinline__ v8f wmma_bf16(v16bf a, v16bf b, v8f c) {
  // args: (neg_a, A, neg_b, B, c_mod, C, reuse_a, reuse_b)
  return __builtin_amdgcn_wmma_f32_16x16x32_bf16(false, a, false, b, (short)0, c, false, false);
}

// ---------------------------------------------------------------------------
// Kernel 1: fp32 -> bf16 conversion (grid-stride)
// ---------------------------------------------------------------------------
__global__ void cvt_bf16_kernel(const float* __restrict__ src,
                                __bf16* __restrict__ dst, int n) {
  int i = blockIdx.x * blockDim.x + threadIdx.x;
  int stride = gridDim.x * blockDim.x;
  for (; i < n; i += stride) dst[i] = (__bf16)src[i];
}

// ---------------------------------------------------------------------------
// Kernel 2: fused QKV projection (+RoPE for Q,K; transposed packed store for V)
// grid = (N/32 token tiles, H heads, 3={Q,K,V}); block = 32 (one wave)
// Each wave: 32 tokens x 64 channels (one full head): 2 M-tiles x 4 N-tiles,
// 8 WMMAs per 32-wide k-step for 6 fragment loads (21.3 FLOP per L2 byte).
// A-frag: lane<16 holds row M=ln, K=k..k+15; lane>=16 same row, K=k+16..k+31.
// B-frag: lane<16 holds col N=ln (= W row h*64+16c+ln), same K split.
// ---------------------------------------------------------------------------
__global__ void qkv_rope_kernel(const __bf16* __restrict__ xb,
                                const __bf16* __restrict__ Wq,
                                const __bf16* __restrict__ Wk,
                                const __bf16* __restrict__ Wv,
                                const int*    __restrict__ tp,
                                __bf16* __restrict__ Qh,   // [B,H,S,DK]
                                __bf16* __restrict__ Kh,   // [B,H,S,DK]
                                __bf16* __restrict__ Vt) { // [B,H,DK,S]
  const int lane = threadIdx.x;
  const int ln = lane & 15, hi = lane >> 4;
  const int tm = blockIdx.x;
  const int h  = blockIdx.y;
  const int which = blockIdx.z;               // 0=Q, 1=K, 2=V
  const __bf16* W = (which == 0) ? Wq : ((which == 1) ? Wk : Wv);
  const int n0 = tm * 32;
  const int bb = n0 / S_;
  const int s0 = n0 % S_;

  v8f acc[2][4] = {};
  const __bf16* arow0 = xb + (size_t)(n0 + ln) * D_ + 16 * hi;
  const __bf16* arow1 = xb + (size_t)(n0 + 16 + ln) * D_ + 16 * hi;
  const __bf16* brow0 = W + (size_t)(h * 64 +  0 + ln) * D_ + 16 * hi;
  const __bf16* brow1 = W + (size_t)(h * 64 + 16 + ln) * D_ + 16 * hi;
  const __bf16* brow2 = W + (size_t)(h * 64 + 32 + ln) * D_ + 16 * hi;
  const __bf16* brow3 = W + (size_t)(h * 64 + 48 + ln) * D_ + 16 * hi;

  for (int k = 0; k < D_; k += 32) {
    v16bf a0 = *(const v16bf*)(arow0 + k);
    v16bf a1 = *(const v16bf*)(arow1 + k);
    v16bf b0 = *(const v16bf*)(brow0 + k);
    v16bf b1 = *(const v16bf*)(brow1 + k);
    v16bf b2 = *(const v16bf*)(brow2 + k);
    v16bf b3 = *(const v16bf*)(brow3 + k);
    acc[0][0] = wmma_bf16(a0, b0, acc[0][0]);
    acc[0][1] = wmma_bf16(a0, b1, acc[0][1]);
    acc[0][2] = wmma_bf16(a0, b2, acc[0][2]);
    acc[0][3] = wmma_bf16(a0, b3, acc[0][3]);
    acc[1][0] = wmma_bf16(a1, b0, acc[1][0]);
    acc[1][1] = wmma_bf16(a1, b1, acc[1][1]);
    acc[1][2] = wmma_bf16(a1, b2, acc[1][2]);
    acc[1][3] = wmma_bf16(a1, b3, acc[1][3]);
  }

  if (which < 2) {
    // RoPE: channel pairs (2p, 2p+1) live in adjacent lanes -> shfl_xor(1).
    __bf16* Out = (which == 0) ? Qh : Kh;
#pragma unroll
    for (int c = 0; c < 4; ++c) {
      const int dd = 16 * c + ln;                         // channel in head
      const float fexp = -(float)(dd & ~1) * (1.0f / 64.0f);
      const float invf = __powf(10000.0f, fexp);          // theta^(-2p/64)
      const float sgn  = (dd & 1) ? 1.0f : -1.0f;
#pragma unroll
      for (int t = 0; t < 2; ++t) {
#pragma unroll
        for (int r = 0; r < 8; ++r) {
          const int s = s0 + 16 * t + r + 8 * hi;
          const float pos = (float)tp[s];
          const float ang = pos * invf;
          const float cs = __cosf(ang), sn = __sinf(ang);
          const float v = acc[t][c][r];
          const float partner = __shfl_xor(v, 1, 32);
          const float res = v * cs + sgn * partner * sn;
          Out[((size_t)(bb * H_ + h) * S_ + s) * DK_ + dd] = (__bf16)res;
        }
      }
    }
  } else {
    // V stored transposed [B,H,DK,S]: 8 consecutive s-positions per lane
    // pack into one 16-byte store.
#pragma unroll
    for (int c = 0; c < 4; ++c) {
      const int dd = 16 * c + ln;
      __bf16* vb = Vt + ((size_t)(bb * H_ + h) * DK_ + dd) * S_ + s0 + 8 * hi;
#pragma unroll
      for (int t = 0; t < 2; ++t) {
        v8bf pack;
#pragma unroll
        for (int r = 0; r < 8; ++r) pack[r] = (__bf16)acc[t][c][r];
        *(v8bf*)(vb + 16 * t) = pack;
      }
    }
  }
}

// ---------------------------------------------------------------------------
// Kernel 3: causal flash attention, online softmax.
// grid = (S/32 query tiles, B*H); block = 32 (one wave, 32 queries).
// Per 32-key tile: 8 QK^T WMMAs + 8 PV WMMAs for 8 K/V fragment loads
// (32 FLOP per L2 byte); K and V fragments are shared across both query
// M-tiles. P goes C-layout -> A-layout through 2KB of LDS.
// ---------------------------------------------------------------------------
__global__ void flash_kernel(const __bf16* __restrict__ Qh,  // [B,H,S,DK]
                             const __bf16* __restrict__ Kh,  // [B,H,S,DK]
                             const __bf16* __restrict__ Vt,  // [B,H,DK,S]
                             __bf16* __restrict__ Oc) {      // [N,D] merged
  __shared__ __bf16 Pl[32 * 32];
  const int lane = threadIdx.x;
  const int ln = lane & 15, hi = lane >> 4;
  const int qt = blockIdx.x;
  const int bh = blockIdx.y;
  const int bb = bh / H_, h = bh % H_;
  const int q0 = qt * 32;

  v16bf aq[2][2];
#pragma unroll
  for (int t = 0; t < 2; ++t) {
    const __bf16* qbase = Qh + ((size_t)bh * S_ + q0 + 16 * t + ln) * DK_;
    aq[t][0] = *(const v16bf*)(qbase + 16 * hi);        // K(dk) 0..31
    aq[t][1] = *(const v16bf*)(qbase + 32 + 16 * hi);   // K(dk) 32..63
  }

  float mst[2][8], lst[2][8];
#pragma unroll
  for (int t = 0; t < 2; ++t)
#pragma unroll
    for (int r = 0; r < 8; ++r) { mst[t][r] = -1e30f; lst[t][r] = 0.0f; }
  v8f o[2][4] = {};

  const int jmax = (q0 + 31) >> 5;
  for (int j = 0; j <= jmax; ++j) {
    const int k0 = j * 32;
    v16bf bk[2][2];
#pragma unroll
    for (int u = 0; u < 2; ++u) {
      const __bf16* kb = Kh + ((size_t)bh * S_ + k0 + 16 * u + ln) * DK_;
      bk[u][0] = *(const v16bf*)(kb + 16 * hi);
      bk[u][1] = *(const v16bf*)(kb + 32 + 16 * hi);
    }

    v8f sc[2][2] = {};
#pragma unroll
    for (int t = 0; t < 2; ++t)
#pragma unroll
      for (int u = 0; u < 2; ++u) {
        sc[t][u] = wmma_bf16(aq[t][0], bk[u][0], sc[t][u]);
        sc[t][u] = wmma_bf16(aq[t][1], bk[u][1], sc[t][u]);
      }

    __syncthreads();   // previous iteration's Pl reads are complete
#pragma unroll
    for (int t = 0; t < 2; ++t) {
#pragma unroll
      for (int r = 0; r < 8; ++r) {
        const int qpos = q0 + 16 * t + r + 8 * hi;
        float v0 = sc[t][0][r] * 0.125f;            // 1/sqrt(64)
        float v1 = sc[t][1][r] * 0.125f;
        if (k0 + ln      > qpos) v0 = -1e30f;       // causal mask
        if (k0 + 16 + ln > qpos) v1 = -1e30f;
        float mx = fmaxf(v0, v1);                   // row reduce, 16 lanes
        mx = fmaxf(mx, __shfl_xor(mx, 1, 32));
        mx = fmaxf(mx, __shfl_xor(mx, 2, 32));
        mx = fmaxf(mx, __shfl_xor(mx, 4, 32));
        mx = fmaxf(mx, __shfl_xor(mx, 8, 32));
        const float mnew  = fmaxf(mst[t][r], mx);
        const float alpha = __expf(mst[t][r] - mnew);
        const float e0 = __expf(v0 - mnew);
        const float e1 = __expf(v1 - mnew);
        float rs = e0 + e1;
        rs += __shfl_xor(rs, 1, 32);
        rs += __shfl_xor(rs, 2, 32);
        rs += __shfl_xor(rs, 4, 32);
        rs += __shfl_xor(rs, 8, 32);
        lst[t][r] = lst[t][r] * alpha + rs;
        mst[t][r] = mnew;
#pragma unroll
        for (int c = 0; c < 4; ++c) o[t][c][r] *= alpha;
        const int row = 16 * t + r + 8 * hi;        // C-layout -> LDS
        Pl[row * 32 + ln]      = (__bf16)e0;
        Pl[row * 32 + 16 + ln] = (__bf16)e1;
      }
    }
    __syncthreads();

    v16bf ap[2];
#pragma unroll
    for (int t = 0; t < 2; ++t)
      ap[t] = *(const v16bf*)(&Pl[(16 * t + ln) * 32 + 16 * hi]);

#pragma unroll
    for (int c = 0; c < 4; ++c) {
      const __bf16* vb = Vt + ((size_t)bh * DK_ + 16 * c + ln) * S_ + k0 + 16 * hi;
      v16bf bv = *(const v16bf*)vb;                 // shared by both M-tiles
      o[0][c] = wmma_bf16(ap[0], bv, o[0][c]);
      o[1][c] = wmma_bf16(ap[1], bv, o[1][c]);
    }
  }

  // Normalize and store merged-head layout [b*S+s, h*64+dd].
#pragma unroll
  for (int t = 0; t < 2; ++t)
#pragma unroll
    for (int c = 0; c < 4; ++c) {
      const int dd = 16 * c + ln;
#pragma unroll
      for (int r = 0; r < 8; ++r) {
        const int s = q0 + 16 * t + r + 8 * hi;
        const float val = o[t][c][r] / lst[t][r];
        Oc[(size_t)(bb * S_ + s) * D_ + h * DK_ + dd] = (__bf16)val;
      }
    }
}

// ---------------------------------------------------------------------------
// Kernel 4: output projection out = O @ Wo^T (fp32 stores to d_out)
// grid = (N/32, D/64); block = 32; 32x64 tile per wave like kernel 2.
// ---------------------------------------------------------------------------
__global__ void outproj_kernel(const __bf16* __restrict__ Oc,
                               const __bf16* __restrict__ Wo,
                               float* __restrict__ out) {
  const int lane = threadIdx.x;
  const int ln = lane & 15, hi = lane >> 4;
  const int tm = blockIdx.x;
  const int h  = blockIdx.y;
  const int n0 = tm * 32;

  v8f acc[2][4] = {};
  const __bf16* arow0 = Oc + (size_t)(n0 + ln) * D_ + 16 * hi;
  const __bf16* arow1 = Oc + (size_t)(n0 + 16 + ln) * D_ + 16 * hi;
  const __bf16* brow0 = Wo + (size_t)(h * 64 +  0 + ln) * D_ + 16 * hi;
  const __bf16* brow1 = Wo + (size_t)(h * 64 + 16 + ln) * D_ + 16 * hi;
  const __bf16* brow2 = Wo + (size_t)(h * 64 + 32 + ln) * D_ + 16 * hi;
  const __bf16* brow3 = Wo + (size_t)(h * 64 + 48 + ln) * D_ + 16 * hi;

  for (int k = 0; k < D_; k += 32) {
    v16bf a0 = *(const v16bf*)(arow0 + k);
    v16bf a1 = *(const v16bf*)(arow1 + k);
    v16bf b0 = *(const v16bf*)(brow0 + k);
    v16bf b1 = *(const v16bf*)(brow1 + k);
    v16bf b2 = *(const v16bf*)(brow2 + k);
    v16bf b3 = *(const v16bf*)(brow3 + k);
    acc[0][0] = wmma_bf16(a0, b0, acc[0][0]);
    acc[0][1] = wmma_bf16(a0, b1, acc[0][1]);
    acc[0][2] = wmma_bf16(a0, b2, acc[0][2]);
    acc[0][3] = wmma_bf16(a0, b3, acc[0][3]);
    acc[1][0] = wmma_bf16(a1, b0, acc[1][0]);
    acc[1][1] = wmma_bf16(a1, b1, acc[1][1]);
    acc[1][2] = wmma_bf16(a1, b2, acc[1][2]);
    acc[1][3] = wmma_bf16(a1, b3, acc[1][3]);
  }

#pragma unroll
  for (int t = 0; t < 2; ++t)
#pragma unroll
    for (int c = 0; c < 4; ++c) {
#pragma unroll
      for (int r = 0; r < 8; ++r) {
        out[(size_t)(n0 + 16 * t + r + 8 * hi) * D_ + h * 64 + 16 * c + ln] =
            acc[t][c][r];
      }
    }
}

// ---------------------------------------------------------------------------
extern "C" void kernel_launch(void* const* d_in, const int* in_sizes, int n_in,
                              void* d_out, int out_size, void* d_ws, size_t ws_size,
                              hipStream_t stream) {
  const float* x  = (const float*)d_in[0];
  const int*   tp = (const int*)d_in[1];
  const float* Wq = (const float*)d_in[2];
  const float* Wk = (const float*)d_in[3];
  const float* Wv = (const float*)d_in[4];
  const float* Wo = (const float*)d_in[5];

  // Workspace layout (bf16), total ~48 MB.
  __bf16* xb  = (__bf16*)d_ws;                   // [N, D]
  __bf16* Wqb = xb  + (size_t)N_ * D_;           // [D, D]
  __bf16* Wkb = Wqb + (size_t)D_ * D_;
  __bf16* Wvb = Wkb + (size_t)D_ * D_;
  __bf16* Wob = Wvb + (size_t)D_ * D_;
  __bf16* Qh  = Wob + (size_t)D_ * D_;           // [B,H,S,DK]
  __bf16* Kh  = Qh  + (size_t)N_ * D_;           // [B,H,S,DK]
  __bf16* Vt  = Kh  + (size_t)N_ * D_;           // [B,H,DK,S]
  __bf16* Oc  = Vt  + (size_t)N_ * D_;           // [N, D]

  cvt_bf16_kernel<<<1024, 256, 0, stream>>>(x,  xb,  N_ * D_);
  cvt_bf16_kernel<<<1024, 256, 0, stream>>>(Wq, Wqb, D_ * D_);
  cvt_bf16_kernel<<<1024, 256, 0, stream>>>(Wk, Wkb, D_ * D_);
  cvt_bf16_kernel<<<1024, 256, 0, stream>>>(Wv, Wvb, D_ * D_);
  cvt_bf16_kernel<<<1024, 256, 0, stream>>>(Wo, Wob, D_ * D_);

  qkv_rope_kernel<<<dim3(N_ / 32, H_, 3), 32, 0, stream>>>(
      xb, Wqb, Wkb, Wvb, tp, Qh, Kh, Vt);

  flash_kernel<<<dim3(S_ / 32, B_ * H_), 32, 0, stream>>>(Qh, Kh, Vt, Oc);

  outproj_kernel<<<dim3(N_ / 32, H_), 32, 0, stream>>>(Oc, Wob, (float*)d_out);
}
